// RNNModel_65506841199167
// MI455X (gfx1250) — matrix operand order repaired and checked by reading, not compiled, not run on hardware
//
#include <hip/hip_runtime.h>
#include <hip/hip_bf16.h>

// ---------------- problem constants ----------------
#define BSZ    64
#define TLEN   512
#define NINP   512
#define NHID   1024
#define G4     4096          // 4*NHID
#define H3     341           // NHID/3
#define MROWS  (TLEN*BSZ)    // 32768

// ---------------- WMMA types / helpers ----------------
typedef __bf16 bf16;
typedef bf16  v16bf __attribute__((ext_vector_type(16)));
typedef float v8f   __attribute__((ext_vector_type(8)));

union FragB { v16bf v; unsigned u[8]; };

// dynamic LDS (sized per-launch)
extern __shared__ unsigned short sh_dyn[];

__device__ __forceinline__ unsigned short f2bf(float x) {
  unsigned u = __float_as_uint(x);
  unsigned r = u + 0x7FFFu + ((u >> 16) & 1u);   // RNE
  return (unsigned short)(r >> 16);
}

// A fragment: 16x32 (MxK) bf16, per ISA 7.12.2 16-bit A layout.
// Works for global or LDS source (addrspace recovered after inlining).
__device__ __forceinline__ v16bf load_a_frag(const unsigned short* X, int ld,
                                             int m0, int k0) {
  int lane = threadIdx.x & 31;
  int half = lane >> 4;
  const unsigned* row = (const unsigned*)(X + (size_t)(m0 + (lane & 15)) * ld + k0);
  FragB f;
#pragma unroll
  for (int v = 0; v < 8; ++v) {
    int kp = ((v & 4) ? 16 : 0) + (v & 3) * 2 + half * 8;
    f.u[v] = row[kp >> 1];
  }
  return f.v;
}

// B fragment: 32x16 (KxN) bf16: lane L holds column N=L%16; VGPR v holds K pair (same K order as A).
__device__ __forceinline__ v16bf load_b_frag(const unsigned short* W, int ld,
                                             int k0, int n0) {
  int lane = threadIdx.x & 31;
  int half = lane >> 4;
  const unsigned short* col = W + (size_t)k0 * ld + n0 + (lane & 15);
  FragB f;
#pragma unroll
  for (int v = 0; v < 8; ++v) {
    int kp = ((v & 4) ? 16 : 0) + (v & 3) * 2 + half * 8;
    f.u[v] = (unsigned)col[(size_t)kp * ld] | ((unsigned)col[(size_t)(kp + 1) * ld] << 16);
  }
  return f.v;
}

__device__ __forceinline__ v8f wmma_bf16(v16bf a, v16bf b, v8f c) {
  return __builtin_amdgcn_wmma_f32_16x16x32_bf16(false, a, false, b,
                                                 (short)0, c, false, false);
}

__device__ __forceinline__ float sigm(float x) {
  return 1.0f / (1.0f + __expf(-x));
}

// async global -> LDS copy of one 16-byte chunk (ASYNCcnt-tracked)
__device__ __forceinline__ void async_copy16(unsigned lds_off,
                                             const void* gaddr) {
  asm volatile("global_load_async_to_lds_b128 %0, %1, off"
               :: "v"(lds_off),
                  "v"((unsigned long long)(uintptr_t)gaddr)
               : "memory");
}
__device__ __forceinline__ void async_wait0() {
  asm volatile("s_wait_asynccnt 0x0" ::: "memory");
}

// ---------------- device-wide barrier (sense via phase counter) ----------------
__device__ __forceinline__ void gsync(unsigned* cnt, volatile unsigned* phase,
                                      unsigned nblk, unsigned& lp) {
  __syncthreads();
  if (threadIdx.x == 0) {
    __threadfence();
    unsigned arr = atomicAdd(cnt, 1u);
    if (arr == nblk - 1u) {
      atomicExch(cnt, 0u);
      __threadfence();
      atomicAdd((unsigned*)phase, 1u);
    } else {
      while (*phase <= lp) { __builtin_amdgcn_s_sleep(1); }
    }
  }
  lp++;
  __syncthreads();
}

// ---------------- kernels ----------------

// init: sync counters, bias = b_ih+b_hh, seed h/c buffers, zero feature acc
__global__ void k_init(unsigned* sync, float* bias,
                       const float* b_ih, const float* b_hh,
                       const float* hx, const float* cx,
                       float* hbuf, float* cbuf, unsigned short* hbf,
                       float* feat) {
  int tid = blockIdx.x * 256 + threadIdx.x;   // 65536 threads
  if (tid < 2) sync[tid] = 0u;
  if (tid < G4) bias[tid] = b_ih[tid] + b_hh[tid];
  float h = hx[tid];
  hbuf[tid] = h;
  hbf[tid]  = f2bf(h);
  cbuf[tid] = cx[tid];
  feat[tid] = 0.0f;
}

// transpose [rows x cols] f32 -> [cols x rows] bf16
__global__ void k_transpose(const float* W, unsigned short* WT, int rows, int cols) {
  size_t total = (size_t)rows * cols;
  for (size_t idx = blockIdx.x * 256ull + threadIdx.x; idx < total;
       idx += (size_t)gridDim.x * 256ull) {
    size_t k = idx / rows;
    size_t n = idx % rows;
    WT[k * rows + n] = f2bf(W[n * cols + k]);
  }
}

// embedding gather: X[(t*B+b)*NINP + i] = bf16(emb_table[input[b*T+t]*NINP + i])
__global__ void k_gather(const int* input, const float* emb, unsigned short* X) {
  size_t total = (size_t)MROWS * NINP;
  for (size_t idx = blockIdx.x * 256ull + threadIdx.x; idx < total;
       idx += (size_t)gridDim.x * 256ull) {
    size_t r = idx >> 9;            // row = t*B + b
    int    i = (int)(idx & 511);
    int    t = (int)(r >> 6);
    int    b = (int)(r & 63);
    int tok = input[b * TLEN + t];
    X[idx] = f2bf(emb[(size_t)tok * NINP + i]);
  }
}

// x_proj GEMM: XP[32768 x 4096] = X[32768 x 512] * WihT + bias
// All 8 waves of a block share one 16-row A tile -> stage it in LDS (16 KB)
// via async global->LDS, then each wave computes a 16x64 output strip.
__global__ void __launch_bounds__(256)
k_xproj(const unsigned short* X, const unsigned short* WT,
        const float* bias, float* XP) {
  int wid = blockIdx.x * 8 + (threadIdx.x >> 5);   // 0..131071
  int mt = wid >> 6;          // 0..2047 (same for all waves in a block)
  int ng = wid & 63;          // 0..63
  int m0 = mt * 16, n0 = ng * 64;

  // stage A tile: 16 rows x 512 cols bf16 = 16 KB = 1024 x 16B chunks
  {
    unsigned base = (unsigned)(uintptr_t)sh_dyn;
    const unsigned short* src = X + (size_t)m0 * NINP;
#pragma unroll
    for (int i = 0; i < 4; ++i) {
      unsigned ch = threadIdx.x + i * 256;      // 0..1023
      async_copy16(base + ch * 16, src + (size_t)ch * 8);
    }
    async_wait0();
    __syncthreads();
  }

  v8f acc0 = {}, acc1 = {}, acc2 = {}, acc3 = {};
  for (int k0 = 0; k0 < NINP; k0 += 32) {
    if (k0 + 32 < NINP)
      __builtin_prefetch(WT + (size_t)(k0 + 32) * G4 + n0, 0, 1);
    v16bf a = load_a_frag(sh_dyn, NINP, 0, k0);   // from LDS
    acc0 = wmma_bf16(a, load_b_frag(WT, G4, k0, n0 +  0), acc0);
    acc1 = wmma_bf16(a, load_b_frag(WT, G4, k0, n0 + 16), acc1);
    acc2 = wmma_bf16(a, load_b_frag(WT, G4, k0, n0 + 32), acc2);
    acc3 = wmma_bf16(a, load_b_frag(WT, G4, k0, n0 + 48), acc3);
  }
  int lane = threadIdx.x & 31, half = lane >> 4, nl = lane & 15;
  v8f accs[4] = {acc0, acc1, acc2, acc3};
#pragma unroll
  for (int j = 0; j < 4; ++j) {
    int n = n0 + j * 16 + nl;
    float bv = bias[n];
#pragma unroll
    for (int v = 0; v < 8; ++v)
      XP[(size_t)(m0 + v + half * 8) * G4 + n] = accs[j][v] + bv;
  }
}

// persistent LSTM scan: 64 blocks x 256 threads (512 waves)
// per step: stage full h (bf16, 128 KB) into LDS via async copy, then
// gates = xproj[t] (as WMMA C operand) + h @ WhhT, grid barrier, fused
// sigmoid/tanh cell update + masked pooling accumulate, grid barrier.
__global__ void __launch_bounds__(256)
k_scan(const float* XP, const unsigned short* WhhT, unsigned short* hbf,
       float* hbuf, float* cbuf, float* gates, float* feat,
       const int* seq_len, unsigned* sync) {
  unsigned lp = 0;
  unsigned* cnt = sync;
  volatile unsigned* phase = sync + 1;
  int wid  = blockIdx.x * 8 + (threadIdx.x >> 5);   // 0..511
  int lane = threadIdx.x & 31, half = lane >> 4, nl = lane & 15;
  int gtid = blockIdx.x * 256 + threadIdx.x;        // 0..16383
  unsigned shbase = (unsigned)(uintptr_t)sh_dyn;

  for (int t = 0; t < TLEN; ++t) {
    const float* xp_t = XP + (size_t)t * BSZ * G4;

    // ---- stage h (64x1024 bf16 = 128 KB = 8192 x 16B chunks) into LDS ----
#pragma unroll
    for (int i = 0; i < 32; ++i) {
      unsigned ch = threadIdx.x + i * 256;          // 0..8191
      async_copy16(shbase + ch * 16, hbf + (size_t)ch * 8);
    }
    async_wait0();
    __syncthreads();

    // ---- GEMM phase: 1024 16x16 tiles over 512 waves -> 2 tiles/wave ----
#pragma unroll
    for (int q8 = 0; q8 < 2; ++q8) {
      int q  = wid + q8 * 512;
      int mt = q >> 8;            // 0..3
      int nt = q & 255;           // 0..255
      int m0 = mt * 16, n0 = nt * 16;
      v8f acc;
#pragma unroll
      for (int v = 0; v < 8; ++v)
        acc[v] = xp_t[(size_t)(m0 + v + half * 8) * G4 + n0 + nl];
      for (int k0 = 0; k0 < NHID; k0 += 32) {
        if (k0 + 32 < NHID)
          __builtin_prefetch(WhhT + (size_t)(k0 + 32) * G4 + n0, 0, 1);
        v16bf a = load_a_frag(sh_dyn, NHID, m0, k0);   // from LDS
        v16bf b = load_b_frag(WhhT, G4, k0, n0);
        acc = wmma_bf16(a, b, acc);
      }
#pragma unroll
      for (int v = 0; v < 8; ++v)
        gates[(size_t)(m0 + v + half * 8) * G4 + n0 + nl] = acc[v];
    }
    gsync(cnt, phase, 64, lp);

    // ---- elementwise LSTM cell + masked pooling: 65536 elems / 16384 thr ----
#pragma unroll
    for (int e = 0; e < 4; ++e) {
      int idx = e * 16384 + gtid;       // 0..65535
      int bb  = idx >> 10;
      int hid = idx & 1023;
      const float* gr = gates + (size_t)bb * G4 + hid;
      float gi = sigm(gr[0]);
      float gf = sigm(gr[NHID]);
      float gg = tanhf(gr[2 * NHID]);
      float go = sigm(gr[3 * NHID]);
      float c  = gf * cbuf[idx] + gi * gg;
      float h  = go * tanhf(c);
      cbuf[idx] = c;
      hbuf[idx] = h;
      hbf[idx]  = f2bf(h);
      if (t < seq_len[bb]) feat[idx] += h;
    }
    gsync(cnt, phase, 64, lp);
  }
}

// fc1: t1[b][j] = fc1_b[j] + sum_k (feat[b][k]/seq_len[b]) * fc1_W[j][k]
__global__ void k_fc1(const float* feat, const int* seq_len,
                      const float* W, const float* b, float* t1) {
  int idx = blockIdx.x * 256 + threadIdx.x;
  if (idx >= BSZ * H3) return;
  int bb = idx / H3, j = idx % H3;
  float inv = 1.0f / (float)seq_len[bb];
  const float* fr = feat + (size_t)bb * NHID;
  const float* wr = W + (size_t)j * NHID;
  float s = b[j];
  for (int k = 0; k < NHID; ++k) s += fr[k] * inv * wr[k];
  t1[bb * H3 + j] = s;
}

// fc2 + log_softmax -> out[0..127]
__global__ void k_fc2(const float* t1, const float* W, const float* b, float* out) {
  int bb = threadIdx.x;
  if (bb >= BSZ) return;
  const float* tr = t1 + (size_t)bb * H3;
  float x0 = b[0], x1 = b[1];
  for (int j = 0; j < H3; ++j) {
    float v = tr[j];
    x0 += v * W[j];
    x1 += v * W[H3 + j];
  }
  float m   = fmaxf(x0, x1);
  float lse = m + logf(__expf(x0 - m) + __expf(x1 - m));
  out[bb * 2 + 0] = x0 - lse;
  out[bb * 2 + 1] = x1 - lse;
}

// final h/c -> out[128 ..]
__global__ void k_copyout(const float* hbuf, const float* cbuf, float* out) {
  int idx = blockIdx.x * 256 + threadIdx.x;   // 131072 threads
  if (idx < BSZ * NHID) out[128 + idx] = hbuf[idx];
  else                  out[128 + idx] = cbuf[idx - BSZ * NHID];
}

// ---------------- launch ----------------
extern "C" void kernel_launch(void* const* d_in, const int* in_sizes, int n_in,
                              void* d_out, int out_size, void* d_ws, size_t ws_size,
                              hipStream_t stream) {
  (void)in_sizes; (void)n_in; (void)out_size; (void)ws_size;
  const int*   input     = (const int*)  d_in[0];
  const float* hx        = (const float*)d_in[1];
  const float* cx        = (const float*)d_in[2];
  const int*   seq_len   = (const int*)  d_in[3];
  const float* emb_table = (const float*)d_in[4];
  const float* W_ih      = (const float*)d_in[5];
  const float* W_hh      = (const float*)d_in[6];
  const float* b_ih      = (const float*)d_in[7];
  const float* b_hh      = (const float*)d_in[8];
  const float* fc1_W     = (const float*)d_in[9];
  const float* fc1_b     = (const float*)d_in[10];
  const float* fc2_W     = (const float*)d_in[11];
  const float* fc2_b     = (const float*)d_in[12];
  float* out = (float*)d_out;

  char* ws = (char*)d_ws;
  size_t off = 0;
  auto take = [&](size_t bytes) {
    size_t r = off;
    off += (bytes + 255) & ~(size_t)255;
    return r;
  };
  size_t o_sync = take(256);
  size_t o_bias = take((size_t)G4 * 4);
  size_t o_wih  = take((size_t)NINP * G4 * 2);       // WihT bf16 [512][4096]
  size_t o_whh  = take((size_t)NHID * G4 * 2);       // WhhT bf16 [1024][4096]
  size_t o_x    = take((size_t)MROWS * NINP * 2);    // X bf16 [32768][512]
  size_t o_xp   = take((size_t)MROWS * G4 * 4);      // x_proj f32 [32768][4096]
  size_t o_h    = take((size_t)BSZ * NHID * 4);
  size_t o_c    = take((size_t)BSZ * NHID * 4);
  size_t o_hbf  = take((size_t)BSZ * NHID * 2);
  size_t o_gate = take((size_t)BSZ * G4 * 4);
  size_t o_feat = take((size_t)BSZ * NHID * 4);
  size_t o_t1   = take((size_t)BSZ * H3 * 4);

  unsigned*       sync = (unsigned*)      (ws + o_sync);
  float*          bias = (float*)         (ws + o_bias);
  unsigned short* wihT = (unsigned short*)(ws + o_wih);
  unsigned short* whhT = (unsigned short*)(ws + o_whh);
  unsigned short* X    = (unsigned short*)(ws + o_x);
  float*          XP   = (float*)         (ws + o_xp);
  float*          hbuf = (float*)         (ws + o_h);
  float*          cbuf = (float*)         (ws + o_c);
  unsigned short* hbf  = (unsigned short*)(ws + o_hbf);
  float*          gate = (float*)         (ws + o_gate);
  float*          feat = (float*)         (ws + o_feat);
  float*          t1   = (float*)         (ws + o_t1);

  k_init     <<<256,   256, 0, stream>>>(sync, bias, b_ih, b_hh, hx, cx,
                                         hbuf, cbuf, hbf, feat);
  k_transpose<<<4096,  256, 0, stream>>>(W_ih, wihT, G4, NINP);
  k_transpose<<<8192,  256, 0, stream>>>(W_hh, whhT, G4, NHID);
  k_gather   <<<8192,  256, 0, stream>>>(input, emb_table, X);
  k_xproj    <<<16384, 256, 16384,  stream>>>(X, wihT, bias, XP);   // 16 KB LDS
  k_scan     <<<64,    256, 131072, stream>>>(XP, whhT, hbf, hbuf, cbuf, gate,
                                              feat, seq_len, sync); // 128 KB LDS
  k_fc1      <<<(BSZ * H3 + 255) / 256, 256, 0, stream>>>(feat, seq_len,
                                                          fc1_W, fc1_b, t1);
  k_fc2      <<<1,     64,  0, stream>>>(t1, fc2_W, fc2_b, out);
  k_copyout  <<<512,   256, 0, stream>>>(hbuf, cbuf, out);
}
